// CoupledLSTM_37744172597251
// MI455X (gfx1250) — matrix operand order repaired and checked by reading, not compiled, not run on hardware
//
#include <hip/hip_runtime.h>
#include <hip/hip_bf16.h>
#include <stdint.h>

// ---------------------------------------------------------------------------
// CoupledLSTM on gfx1250 (MI455X)
//   Phase 1: xi/xc/xo = x @ W_x^T  via v_wmma_f32_16x16x32_bf16, 3 gates fused
//            per wave (A-tile loaded once, reused for all three gate GEMMs)
//   Phase 2: persistent recurrent kernel, W_h slices pinned in LDS,
//            cell state held in WMMA accumulator registers, grid barrier/step
// ---------------------------------------------------------------------------

typedef __attribute__((ext_vector_type(16))) __bf16   v16bf;
typedef __attribute__((ext_vector_type(8)))  float    v8f;
typedef __attribute__((ext_vector_type(4)))  uint32_t u32x4;

union FragBF {
  v16bf v;
  u32x4 u[2];
};

static constexpr int S = 512, B = 64, I = 512, H = 512;
static constexpr int SB = S * B;                      // 32768 rows
static constexpr int MT = SB / 16;                    // 2048 M tiles
static constexpr int NT = H / 16;                     // 32 N tiles
static constexpr int NBLK = H / 16;                   // 32 persistent blocks
static constexpr size_t NXE = (size_t)SB * I;         // x elements
static constexpr size_t NWE = (size_t)H * I;          // one weight matrix
static constexpr size_t GST = (size_t)S * B * H;      // per-gate preact elems
static constexpr int LDSW = 520;                      // padded LDS row stride (bf16)

__device__ __forceinline__ unsigned short f32_to_bf16(float f) {
  uint32_t u = __float_as_uint(f);
  uint32_t r = 0x7FFFu + ((u >> 16) & 1u);            // round-to-nearest-even
  return (unsigned short)((u + r) >> 16);
}
__device__ __forceinline__ float fast_sigmoid(float x) {
  return 1.0f / (1.0f + __expf(-x));                  // v_exp_f32 + rcp
}
__device__ __forceinline__ float fast_tanh(float x) {
  return 2.0f / (1.0f + __expf(-2.0f * x)) - 1.0f;
}

// ---------------------------------------------------------------------------
// Kernel 0: one-time f32 -> bf16 conversion of x and the six weight matrices,
// plus zero-init of the h ping-pong buffers and the grid-barrier counter.
// ---------------------------------------------------------------------------
__global__ void __launch_bounds__(256) convert_init_kernel(
    const float* __restrict__ x,
    const float* __restrict__ wxi, const float* __restrict__ whi,
    const float* __restrict__ wxc, const float* __restrict__ whc,
    const float* __restrict__ wxo, const float* __restrict__ who,
    unsigned short* __restrict__ xb,
    unsigned short* __restrict__ wb,     // 6 packed H*I bf16 matrices
    unsigned short* __restrict__ hb0,
    unsigned short* __restrict__ hb1,
    unsigned int*   __restrict__ bar) {
  const size_t tid    = (size_t)blockIdx.x * blockDim.x + threadIdx.x;
  const size_t stride = (size_t)gridDim.x * blockDim.x;
  for (size_t i = tid; i < NXE; i += stride) xb[i] = f32_to_bf16(x[i]);
  for (size_t i = tid; i < NWE; i += stride) {
    wb[0 * NWE + i] = f32_to_bf16(wxi[i]);
    wb[1 * NWE + i] = f32_to_bf16(whi[i]);
    wb[2 * NWE + i] = f32_to_bf16(wxc[i]);
    wb[3 * NWE + i] = f32_to_bf16(whc[i]);
    wb[4 * NWE + i] = f32_to_bf16(wxo[i]);
    wb[5 * NWE + i] = f32_to_bf16(who[i]);
  }
  const size_t NH = (size_t)B * H;
  for (size_t i = tid; i < NH; i += stride) { hb0[i] = 0; hb1[i] = 0; }
  if (tid == 0) *bar = 0u;
}

// ---------------------------------------------------------------------------
// Kernel 1: gate pre-activations  xg[g][s][b][h] = (x @ W_xg^T)
// One wave per (mt, nt) tile, all THREE gates fused: A-frag loaded once per
// K-step and reused for 3 WMMAs (8 b128 loads : 3 wmma instead of 12 : 3).
// A/B fragments use contiguous K (k = 16*hi + s); D layout follows CDNA5 ISA §7.12.2.
// ---------------------------------------------------------------------------
__global__ void __launch_bounds__(256) gate_proj_kernel(
    const unsigned short* __restrict__ xb,
    const unsigned short* __restrict__ wb,
    float* __restrict__ xg) {
  const int lane = threadIdx.x & 31;
  const int wave = threadIdx.x >> 5;
  const int tile = blockIdx.x * 8 + wave;               // 0 .. MT*NT-1
  const int mt = tile >> 5;                              // M tile (row/16)
  const int nt = tile & 31;                              // N tile (col/16)
  const int nl = lane & 15;
  const int hi = lane >> 4;

  // A: row m = nl of the 16-row strip; B: W_x row (h) = nl of the 16-col strip
  const unsigned short* arow  = xb + ((size_t)mt * 16 + nl) * I;
  const size_t          wrow  = ((size_t)nt * 16 + nl) * I;
  const unsigned short* brow0 = wb + 0 * NWE + wrow;     // W_xi
  const unsigned short* brow1 = wb + 2 * NWE + wrow;     // W_xc
  const unsigned short* brow2 = wb + 4 * NWE + wrow;     // W_xo

  v8f ai = {}, ac = {}, ao = {};
#pragma unroll
  for (int kk = 0; kk < 16; ++kk) {
    FragBF a, b0, b1, b2;
    const int ka = kk * 32 + 16 * hi;       // A: 16 contiguous K per half-wave
    a.u[0] = *(const u32x4*)(arow + ka);
    a.u[1] = *(const u32x4*)(arow + ka + 8);
    const int kb = kk * 32 + 16 * hi;       // B: 16 contiguous K per half-wave
    b0.u[0] = *(const u32x4*)(brow0 + kb);
    b0.u[1] = *(const u32x4*)(brow0 + kb + 8);
    b1.u[0] = *(const u32x4*)(brow1 + kb);
    b1.u[1] = *(const u32x4*)(brow1 + kb + 8);
    b2.u[0] = *(const u32x4*)(brow2 + kb);
    b2.u[1] = *(const u32x4*)(brow2 + kb + 8);
    ai = __builtin_amdgcn_wmma_f32_16x16x32_bf16(false, a.v, false, b0.v,
                                                 (short)0, ai, false, false);
    ac = __builtin_amdgcn_wmma_f32_16x16x32_bf16(false, a.v, false, b1.v,
                                                 (short)0, ac, false, false);
    ao = __builtin_amdgcn_wmma_f32_16x16x32_bf16(false, a.v, false, b2.v,
                                                 (short)0, ao, false, false);
  }

  // D layout: lane -> col n = nl; VGPR v -> row M = v + 8*hi
  float* obase = xg + ((size_t)mt * 16) * H + (size_t)nt * 16;
#pragma unroll
  for (int v = 0; v < 8; ++v) {
    const size_t off = (size_t)(8 * hi + v) * H + nl;
    obase[0 * GST + off] = ai[v];
    obase[1 * GST + off] = ac[v];
    obase[2 * GST + off] = ao[v];
  }
}

// ---------------------------------------------------------------------------
// Kernel 2: persistent recurrent scan. 32 blocks x 128 threads (4 waves).
// Block owns H-slice [h0, h0+16). Wave w owns batch rows [16w, 16w+16).
// W_hi/W_hc/W_ho slices live in LDS; cell state lives in v8f accumulators.
// h ping-pongs between two bf16 global buffers with a grid barrier per step.
// ---------------------------------------------------------------------------
__global__ void __launch_bounds__(128) lstm_scan_kernel(
    const unsigned short* __restrict__ wb,
    const float* __restrict__ xg,
    const float* __restrict__ b_i,
    const float* __restrict__ b_c,
    const float* __restrict__ b_o,
    unsigned short* __restrict__ hb0,
    unsigned short* __restrict__ hb1,
    float* __restrict__ out,
    unsigned int* __restrict__ bar) {
  __shared__ unsigned short ldsW[3][16][LDSW];   // ~49.9 KB of 320 KB WGP pool

  const int tid = threadIdx.x;
  const int h0  = blockIdx.x * 16;

  // Stage recurrent weight slices into LDS (bf16). wb slot g*2+1 = W_h of gate g.
  for (int idx = tid; idx < 3 * 16 * 512; idx += 128) {
    const int g = idx / (16 * 512);
    const int r = (idx >> 9) & 15;
    const int k = idx & 511;
    ldsW[g][r][k] = wb[(size_t)(g * 2 + 1) * NWE + (size_t)(h0 + r) * I + k];
  }
  __syncthreads();

  const int wave = tid >> 5, lane = tid & 31;
  const int nl = lane & 15, hi = lane >> 4;
  const int m0 = wave * 16;                 // batch-row base for this wave
  const int gn = h0 + nl;                   // global hidden column for this lane
  const float bi = b_i[gn], bc = b_c[gn], bo = b_o[gn];

  v8f cst = {};                             // cell state, WMMA C/D layout

  unsigned short* hbR = hb0;                // read h_{t-1}
  unsigned short* hbW = hb1;                // write h_t

#pragma unroll 1
  for (int s = 0; s < S; ++s) {
    v8f ai = {}, ac = {}, ao = {};
    const unsigned short* arow = hbR + (size_t)(m0 + nl) * H;
#pragma unroll
    for (int kk = 0; kk < 16; ++kk) {
      FragBF a, b0, b1, b2;
      const int ka = kk * 32 + 16 * hi;
      a.u[0] = *(const u32x4*)(arow + ka);
      a.u[1] = *(const u32x4*)(arow + ka + 8);
      const int kb = kk * 32 + 16 * hi;
      b0.u[0] = *(const u32x4*)(&ldsW[0][nl][kb]);
      b0.u[1] = *(const u32x4*)(&ldsW[0][nl][kb + 8]);
      b1.u[0] = *(const u32x4*)(&ldsW[1][nl][kb]);
      b1.u[1] = *(const u32x4*)(&ldsW[1][nl][kb + 8]);
      b2.u[0] = *(const u32x4*)(&ldsW[2][nl][kb]);
      b2.u[1] = *(const u32x4*)(&ldsW[2][nl][kb + 8]);
      ai = __builtin_amdgcn_wmma_f32_16x16x32_bf16(false, a.v, false, b0.v,
                                                   (short)0, ai, false, false);
      ac = __builtin_amdgcn_wmma_f32_16x16x32_bf16(false, a.v, false, b1.v,
                                                   (short)0, ac, false, false);
      ao = __builtin_amdgcn_wmma_f32_16x16x32_bf16(false, a.v, false, b2.v,
                                                   (short)0, ao, false, false);
    }

    const float* xgi = xg + 0 * GST + (size_t)s * (B * H);
    const float* xgc = xg + 1 * GST + (size_t)s * (B * H);
    const float* xgo = xg + 2 * GST + (size_t)s * (B * H);
    float* outs = out + (size_t)s * (B * H);

#pragma unroll
    for (int v = 0; v < 8; ++v) {
      const int M = m0 + 8 * hi + v;                  // batch row
      const size_t off = (size_t)M * H + gn;
      const float it = fast_sigmoid(ai[v] + xgi[off] + bi);
      const float ch = fast_tanh  (ac[v] + xgc[off] + bc);
      const float ot = fast_sigmoid(ao[v] + xgo[off] + bo);
      const float cn = (1.0f - it) * cst[v] + it * ch;  // coupled forget gate
      cst[v] = cn;
      const float hv = ot * fast_tanh(cn);
      outs[off] = hv;
      hbW[off] = f32_to_bf16(hv);
      if (s == S - 1) {
        out[(size_t)S * B * H + off]                 = hv;   // h_t
        out[(size_t)S * B * H + (size_t)B * H + off] = cn;   // c_t
      }
    }

    // ---- grid barrier: all h_t writes visible before next step's GEMM ----
    __threadfence();
    __syncthreads();
    if (tid == 0) {
      __hip_atomic_fetch_add(bar, 1u, __ATOMIC_RELEASE, __HIP_MEMORY_SCOPE_AGENT);
      const unsigned int target = (unsigned int)(NBLK * (s + 1));
      while (__hip_atomic_load(bar, __ATOMIC_ACQUIRE, __HIP_MEMORY_SCOPE_AGENT) < target) {
        __builtin_amdgcn_s_sleep(1);
      }
    }
    __syncthreads();
    __threadfence();

    unsigned short* t = hbR; hbR = hbW; hbW = t;        // ping-pong
  }
}

// ---------------------------------------------------------------------------
// Host-side launcher
// ---------------------------------------------------------------------------
extern "C" void kernel_launch(void* const* d_in, const int* in_sizes, int n_in,
                              void* d_out, int out_size, void* d_ws, size_t ws_size,
                              hipStream_t stream) {
  const float* x    = (const float*)d_in[0];
  const float* wxi  = (const float*)d_in[1];
  const float* whi  = (const float*)d_in[2];
  const float* b_i  = (const float*)d_in[3];
  const float* wxc  = (const float*)d_in[4];
  const float* whc  = (const float*)d_in[5];
  const float* b_c  = (const float*)d_in[6];
  const float* wxo  = (const float*)d_in[7];
  const float* who  = (const float*)d_in[8];
  const float* b_o  = (const float*)d_in[9];
  float* out = (float*)d_out;

  // Workspace layout (bytes), all 256-aligned:
  //   xb   : SB*I bf16                    = 33,554,432
  //   wb   : 6 * H*I bf16                 =  3,145,728
  //   xg   : 3 * S*B*H f32                = 201,326,592
  //   hb0/1: B*H bf16 each                =     65,536 x2
  //   bar  : grid-barrier counter         =        256
  char* ws = (char*)d_ws;
  unsigned short* xb  = (unsigned short*)(ws);
  unsigned short* wb  = (unsigned short*)(ws + 33554432);
  float*          xg  = (float*)        (ws + 36700160);
  unsigned short* hb0 = (unsigned short*)(ws + 238026752);
  unsigned short* hb1 = (unsigned short*)(ws + 238092288);
  unsigned int*   bar = (unsigned int*)  (ws + 238157824);

  convert_init_kernel<<<2048, 256, 0, stream>>>(x, wxi, whi, wxc, whc, wxo, who,
                                                xb, wb, hb0, hb1, bar);

  // 2048 M-tiles * 32 N-tiles = 65,536 tiles (3 gates fused); 8 waves / block
  gate_proj_kernel<<<(MT * NT) / 8, 256, 0, stream>>>(xb, wb, xg);

  lstm_scan_kernel<<<NBLK, 128, 0, stream>>>(wb, xg, b_i, b_c, b_o,
                                             hb0, hb1, out, bar);
}
